// DRELoss_neg_38319698215597
// MI455X (gfx1250) — matrix operand order repaired and checked by reading, not compiled, not run on hardware
//
#include <hip/hip_runtime.h>
#include <hip/hip_bf16.h>

// Shapes (fixed by reference): b=64, C=20, d=768, H=W=64, HW=4096
#define NB    64
#define NC    20
#define ND    768
#define NHW   4096
#define KSTEPS (ND / 32)        // 24 k-steps of K=32
#define NBLK  128               // pixels per block (8 waves x 16)
#define NGRIDX (NHW / NBLK)     // 32

typedef __attribute__((ext_vector_type(16))) __bf16 v16bf;
typedef __attribute__((ext_vector_type(8)))  float  v8f;
typedef __attribute__((ext_vector_type(4)))  float  f32x4;

static __device__ __forceinline__ unsigned short f2bf(float f) {
  unsigned int u = __float_as_uint(f);
  unsigned int r = u + 0x7FFFu + ((u >> 16) & 1u);   // round-to-nearest-even
  return (unsigned short)(r >> 16);
}

// pack two f32 -> {lo: bf16(a), hi: bf16(b)} in one dword.
// Fast path: round-half-up then one v_perm_b32 selecting the two high halves:
//   result bytes [b3 b2 b1 b0] = [ub.b3 ub.b2 ua.b3 ua.b2]  -> sel 0x07060302
static __device__ __forceinline__ unsigned int pack2bf(float a, float b) {
  unsigned int ua = __float_as_uint(a) + 0x8000u;
  unsigned int ub = __float_as_uint(b) + 0x8000u;
  return __builtin_amdgcn_perm(ub, ua, 0x07060302u);
}

// ---------------------------------------------------------------------------
// Kernel 1: L2-normalize cls tokens and pack into WMMA bf16 A-matrix layout.
// Packed layout: PA[b][tile(2)][ks(24)][lane(32)][j(16)] (ushort/bf16)
//   lane = h*16 + m (m = row in 16-row tile, class c = tile*16+m)
//   k_local = h*16 + j;  k = ks*32 + k_local
// Classes >= 20 are zero padding.
// ---------------------------------------------------------------------------
__global__ void dre_normpack(const float* __restrict__ cls,
                             unsigned short* __restrict__ packedA) {
  const int blk = blockIdx.x;         // 0..127
  const int bb  = blk >> 1;
  const int t   = blk & 1;
  const int tid = threadIdx.x;        // 0..255

  __shared__ float partial[256];
  __shared__ float rnorm[16];

  const int row = tid >> 4;           // 0..15 (row within tile)
  const int sub = tid & 15;
  const int c   = t * 16 + row;

  float ss = 0.f;
  if (c < NC) {
    const float* p = cls + ((size_t)bb * NC + c) * ND + sub * (ND / 16);
    #pragma unroll 8
    for (int i = 0; i < ND / 16; ++i) { float x = p[i]; ss += x * x; }
  }
  partial[tid] = ss;
  __syncthreads();
  if (sub == 0) {
    float s = 0.f;
    #pragma unroll
    for (int i = 0; i < 16; ++i) s += partial[row * 16 + i];
    rnorm[row] = 1.f / fmaxf(sqrtf(s), 1e-8f);
  }
  __syncthreads();

  unsigned short* out = packedA + ((size_t)bb * 2 + t) * (KSTEPS * 512);
  // 24*512 = 12288 outputs per block -> 48 per thread
  for (int i = 0; i < 48; ++i) {
    int idx  = tid + i * 256;
    int ks   = idx >> 9;
    int rem  = idx & 511;
    int lane = rem >> 4;
    int j    = rem & 15;
    int h    = lane >> 4;
    int m    = lane & 15;
    int kl   = h * 16 + j;
    int k    = ks * 32 + kl;
    int cc   = t * 16 + m;
    float v  = 0.f;
    if (cc < NC) v = cls[((size_t)bb * NC + cc) * ND + k] * rnorm[m];
    out[idx] = f2bf(v);
  }
}

// ---------------------------------------------------------------------------
// Kernel 2: main fused GEMM (bf16 WMMA) + exp/masked reductions.
// grid = (32, 64); block = 256 (8 waves). Each block: one batch, 128 pixels.
// Double-buffered LDS tile -> single barrier per K-iteration.
// ---------------------------------------------------------------------------
__device__ __forceinline__ void dre_load_conv(const float* __restrict__ pbase,
                                              int ks, int tid,
                                              uint4& d0, uint4& d1) {
  uint4* outs[2] = { &d0, &d1 };
  #pragma unroll
  for (int i = 0; i < 2; ++i) {
    int task = tid + i * 256;                // 0..511
    int kp   = task >> 5;                    // 0..15 (K pair index)
    int n0   = (task & 31) << 2;             // 0..124
    const float* p0 = pbase + ((size_t)ks * 32 + kp * 2) * NHW + n0;
    f32x4 r0 = __builtin_nontemporal_load((const f32x4*)p0);          // k even
    f32x4 r1 = __builtin_nontemporal_load((const f32x4*)(p0 + NHW));  // k odd
    uint4 d;
    d.x = pack2bf(r0.x, r1.x);
    d.y = pack2bf(r0.y, r1.y);
    d.z = pack2bf(r0.z, r1.z);
    d.w = pack2bf(r0.w, r1.w);
    *outs[i] = d;
  }
}

__global__ void __launch_bounds__(256)
dre_main(const float* __restrict__ patch,
         const unsigned short* __restrict__ packedA,
         const int* __restrict__ masks,
         float* __restrict__ sumExp,
         float* __restrict__ flagS) {
  const int bb     = blockIdx.y;
  const int nblock = blockIdx.x;
  const int tid    = threadIdx.x;
  const int wave   = tid >> 5;
  const int lane   = tid & 31;
  const int h      = lane >> 4;
  const int nl     = lane & 15;

  // Double-buffered bf16 K-pair-packed tile: dword (kpair,n) = {k=2p lo, k=2p+1 hi}
  __shared__ uint4 ldsB4[2][512];            // 2 x (16 kpairs x 128 n) dwords

  const float* pbase = patch + (size_t)bb * ND * NHW + nblock * NBLK;
  const unsigned short* aBase0 = packedA + ((size_t)bb * 2 + 0) * (KSTEPS * 512);
  const unsigned short* aBase1 = packedA + ((size_t)bb * 2 + 1) * (KSTEPS * 512);

  v8f acc0 = {}; v8f acc1 = {};

  // prologue: fill buffer 0
  {
    uint4 d0, d1;
    dre_load_conv(pbase, 0, tid, d0, d1);
    ldsB4[0][tid] = d0;
    ldsB4[0][tid + 256] = d1;
  }

  for (int ks = 0; ks < KSTEPS; ++ks) {
    const int cur = ks & 1;
    __syncthreads();                         // lds[cur] ready; lds[1-cur] free

    // issue next tile's global loads early (overlap with WMMA below)
    uint4 e0, e1;
    const bool more = (ks + 1 < KSTEPS);
    if (more) dre_load_conv(pbase, ks + 1, tid, e0, e1);

    // A fragments (pre-packed, contiguous 32B per lane)
    union AF { v16bf v; uint4 q[2]; } a0, a1;
    const uint4* ap0 = (const uint4*)(aBase0 + (size_t)ks * 512 + lane * 16);
    const uint4* ap1 = (const uint4*)(aBase1 + (size_t)ks * 512 + lane * 16);
    a0.q[0] = ap0[0]; a0.q[1] = ap0[1];
    a1.q[0] = ap1[0]; a1.q[1] = ap1[1];

    // B fragment: lane (h,nl) -> column n = wave*16+nl; VGPR v -> kpair h*8+v
    union BF { v16bf v; unsigned int u[8]; } bf;
    const unsigned int* ldsW = (const unsigned int*)&ldsB4[cur][0];
    const int ncol = wave * 16 + nl;
    #pragma unroll
    for (int v = 0; v < 8; ++v) bf.u[v] = ldsW[(h * 8 + v) * NBLK + ncol];

    acc0 = __builtin_amdgcn_wmma_f32_16x16x32_bf16(false, a0.v, false, bf.v,
                                                   (short)0, acc0, false, false);
    acc1 = __builtin_amdgcn_wmma_f32_16x16x32_bf16(false, a1.v, false, bf.v,
                                                   (short)0, acc1, false, false);

    if (more) {
      ldsB4[1 - cur][tid] = e0;
      ldsB4[1 - cur][tid + 256] = e1;
    }
  }

  // Epilogue: acc layout (f32 C/D): VGPR r, lanes0-15 M=r, lanes16-31 M=8+r,
  // N = lane%16. Our column n = nblock*128 + wave*16 + nl.
  const int n    = nblock * NBLK + wave * 16 + nl;
  const int mval = masks[bb * NHW + n];
  float* seB = sumExp + bb * NC;
  float* fsB = flagS  + bb * NC;

  #pragma unroll
  for (int r = 0; r < 8; ++r) {
    { // tile 0: classes 0..15
      int c   = h * 8 + r;
      float s = acc0[r];
      float e = __expf(s);
      float fs = (mval == c + 1) ? s : 0.f;
      #pragma unroll
      for (int off = 1; off < 16; off <<= 1) {
        e  += __shfl_xor(e,  off, 16);
        fs += __shfl_xor(fs, off, 16);
      }
      if (nl == 0) {
        atomicAdd(&seB[c], e);
        atomicAdd(&fsB[c], fs);
      }
    }
    { // tile 1: classes 16..19 (rest is padding)
      int c = 16 + h * 8 + r;
      if (c < NC) {
        float s = acc1[r];
        float e = __expf(s);
        float fs = (mval == c + 1) ? s : 0.f;
        #pragma unroll
        for (int off = 1; off < 16; off <<= 1) {
          e  += __shfl_xor(e,  off, 16);
          fs += __shfl_xor(fs, off, 16);
        }
        if (nl == 0) {
          atomicAdd(&seB[c], e);
          atomicAdd(&fsB[c], fs);
        }
      }
    }
  }
}

// ---------------------------------------------------------------------------
// Kernel 3: per-batch mask histogram -> flagCount[b][c]
// ---------------------------------------------------------------------------
__global__ void dre_hist(const int* __restrict__ masks,
                         float* __restrict__ flagCount) {
  __shared__ int bins[NC + 1];
  const int bb  = blockIdx.x;
  const int tid = threadIdx.x;
  if (tid <= NC) bins[tid] = 0;
  __syncthreads();
  const int* m = masks + bb * NHW;
  for (int i = tid; i < NHW; i += 256) {
    int v = m[i];
    if (v >= 1 && v <= NC) atomicAdd(&bins[v], 1);
  }
  __syncthreads();
  if (tid < NC) flagCount[bb * NC + tid] = (float)bins[tid + 1];
}

// ---------------------------------------------------------------------------
// Kernel 4: final reduction over 1280 (b,c) pairs -> scalar loss
// ---------------------------------------------------------------------------
__global__ void dre_finalize(const int* __restrict__ labels,
                             const float* __restrict__ sumExp,
                             const float* __restrict__ flagS,
                             const float* __restrict__ flagCount,
                             float* __restrict__ out) {
  __shared__ float partial[256];
  const int tid = threadIdx.x;
  float acc = 0.f;
  for (int i = tid; i < NB * NC; i += 256) {
    if (labels[i] == 1) {
      float fc = flagCount[i];
      float per = (fc * logf(sumExp[i] + 1e-4f) - flagS[i]) / (fc + 1e-4f);
      acc += per;
    }
  }
  partial[tid] = acc;
  __syncthreads();
  for (int s = 128; s > 0; s >>= 1) {
    if (tid < s) partial[tid] += partial[tid + s];
    __syncthreads();
  }
  if (tid == 0) out[0] = partial[0] / (float)NB;
}

extern "C" void kernel_launch(void* const* d_in, const int* in_sizes, int n_in,
                              void* d_out, int out_size, void* d_ws, size_t ws_size,
                              hipStream_t stream) {
  (void)in_sizes; (void)n_in; (void)out_size; (void)ws_size;
  const float* cls    = (const float*)d_in[0];   // [64,20,768]
  const float* patch  = (const float*)d_in[1];   // [64,768,64,64]
  const int*   masks  = (const int*)d_in[2];     // [64,64,64]
  const int*   labels = (const int*)d_in[3];     // [64,20]
  float* out = (float*)d_out;

  char* ws = (char*)d_ws;
  const size_t packedA_bytes = (size_t)NB * 2 * KSTEPS * 512 * sizeof(unsigned short); // 3,145,728
  unsigned short* packedA  = (unsigned short*)ws;
  float* sumExp    = (float*)(ws + packedA_bytes);
  float* flagS     = sumExp + NB * NC;
  float* flagCount = flagS  + NB * NC;

  // zero the atomic accumulators (sumExp + flagS are contiguous)
  hipMemsetAsync(sumExp, 0, (size_t)2 * NB * NC * sizeof(float), stream);

  dre_normpack<<<NB * 2, 256, 0, stream>>>(cls, packedA);
  dre_main<<<dim3(NGRIDX, NB), 256, 0, stream>>>(patch, packedA, masks, sumExp, flagS);
  dre_hist<<<NB, 256, 0, stream>>>(masks, flagCount);
  dre_finalize<<<1, 256, 0, stream>>>(labels, sumExp, flagS, flagCount, out);
}
